// MultiHeadAttentionLayer_23630910062795
// MI455X (gfx1250) — compile-verified, matrix operand order checked by reading, not run on hardware
//
#include <hip/hip_runtime.h>

// Problem constants (fixed by the reference).
#define EMBED  768
#define HEADS  12
#define HSIZE  64
#define BATCH  2
#define SEQ    2048
#define MTOK   (BATCH * SEQ)   // 4096 rows for the projections

typedef __attribute__((ext_vector_type(16))) _Float16 v16h;
typedef __attribute__((ext_vector_type(8)))  _Float16 v8h;
typedef __attribute__((ext_vector_type(4)))  _Float16 v4h;
typedef __attribute__((ext_vector_type(8)))  float    v8f;

// D = A(16x32 f16) * B(32x16 f16) + C(16x16 f32)
__device__ __forceinline__ v8f wmma_f16(v16h a, v16h b, v8f c) {
  return __builtin_amdgcn_wmma_f32_16x16x32_f16(
      /*neg_a=*/false, a, /*neg_b=*/false, b,
      /*c_mod=*/(short)0, c, /*reuse_a=*/false, /*reuse_b=*/false);
}

// Async copy of 16 bytes/lane global -> LDS (ASYNCcnt-tracked, no VGPR round trip).
__device__ __forceinline__ void async_b128(const _Float16* gptr, const _Float16* lptr) {
  const unsigned long long ga = (unsigned long long)(uintptr_t)gptr;
  const unsigned la = (unsigned)(uintptr_t)lptr;   // low 32 bits = LDS byte offset
  asm volatile("global_load_async_to_lds_b128 %0, %1, off"
               :: "v"(la), "v"(ga) : "memory");
}
// Wait until at most N async ops remain in flight (in-order completion ->
// waiting <=N with N ops just issued drains everything older).
template <int N>
__device__ __forceinline__ void wait_async_le() {
  asm volatile("s_wait_asynccnt %0" :: "i"(N) : "memory");
}

// Load one 16x32 f16 A-layout fragment from a row-major [rows][stride] array.
// ISA layout: lanes 0-15 hold K={kb..kb+7, kb+16..kb+23} of row (lane&15),
// lanes 16-31 get kb=8. Two 16-byte loads per lane.
__device__ __forceinline__ v16h frag_ld(const _Float16* base, int stride,
                                        int row, int kb) {
  const _Float16* p = base + row * stride + kb;
  v8h lo = *(const v8h*)(p);
  v8h hi = *(const v8h*)(p + 16);
  v16h r;
#pragma unroll
  for (int i = 0; i < 8; ++i) { r[i] = lo[i]; r[i + 8] = hi[i]; }
  return r;
}

// Elementwise f32 -> f16 (vectorized x4); n must be a multiple of 4.
__global__ __launch_bounds__(256)
void cvt_f16(const float* __restrict__ in, _Float16* __restrict__ out, int n4) {
  const int i = blockIdx.x * 256 + threadIdx.x;
  if (i < n4) {
    const float4 f = ((const float4*)in)[i];
    v4h h;
    h[0] = (_Float16)f.x; h[1] = (_Float16)f.y;
    h[2] = (_Float16)f.z; h[3] = (_Float16)f.w;
    ((v4h*)out)[i] = h;
  }
}

// C = A[M,K](f16) * Bt[N,K]^T(f16) + bias[N], M=4096, N=768, K=768.
// 64x64 tiles, BK=64, double-buffered async-to-LDS staging (DMA overlaps WMMA).
// mode 0: out f16 [B,H,S,64]   (Q/K head layout; outScale folds 1/sqrt(d))
// mode 1: out f16 [B,H,64,S]   (V transposed so it feeds the WMMA B operand)
// mode 2: out f32 [M,768]      (final projection into d_out)
__global__ __launch_bounds__(256)
void gemm_wmma(const _Float16* __restrict__ A, const _Float16* __restrict__ Bt,
               const float* __restrict__ bias, void* __restrict__ outp,
               int mode, float outScale) {
  __shared__ _Float16 As[2][64 * 64];
  __shared__ _Float16 Bs[2][64 * 64];

  const int tid   = threadIdx.x;
  const int wave  = tid >> 5;
  const int lane  = tid & 31;
  const int l16   = lane & 15;
  const int lhalf = lane >> 4;
  const int kb    = lhalf ? 8 : 0;
  const int mw    = wave & 3;    // m-subtile 0..3
  const int nw    = wave >> 2;   // n-half 0..1
  const int m0    = blockIdx.y * 64;
  const int n0    = blockIdx.x * 64;
  const int K     = EMBED;
  const int NIT   = K / 64;      // 12

  // Staging: tile = 64 rows x 128B = 512 x 16B chunks; each thread moves 2 per tile.
  const int row0 = tid >> 3;          // 0..31
  const int row1 = row0 + 32;         // 32..63
  const int seg  = (tid & 7) * 8;     // halves within row

  auto stage = [&](int kc, int buf) {
    async_b128(A  + (m0 + row0) * K + kc + seg, &As[buf][row0 * 64 + seg]);
    async_b128(A  + (m0 + row1) * K + kc + seg, &As[buf][row1 * 64 + seg]);
    async_b128(Bt + (n0 + row0) * K + kc + seg, &Bs[buf][row0 * 64 + seg]);
    async_b128(Bt + (n0 + row1) * K + kc + seg, &Bs[buf][row1 * 64 + seg]);
  };

  v8f acc[2] = {{}, {}};

  stage(0, 0);   // prologue: tile 0 in flight
#pragma unroll 1
  for (int it = 0; it < NIT; ++it) {
    const int cur = it & 1;
    if (it + 1 < NIT) {
      stage((it + 1) * 64, cur ^ 1);   // next tile streams during this compute
      wait_async_le<4>();              // drain tile `it` only
    } else {
      wait_async_le<0>();
    }
    __syncthreads();

    const int arow = 16 * mw + l16;
    const v16h af0 = frag_ld(As[cur],      64, arow, kb);   // k 0..31
    const v16h af1 = frag_ld(As[cur] + 32, 64, arow, kb);   // k 32..63
#pragma unroll
    for (int j = 0; j < 2; ++j) {
      const int brow = 32 * nw + 16 * j + l16;
      const v16h bf0 = frag_ld(Bs[cur],      64, brow, kb);
      const v16h bf1 = frag_ld(Bs[cur] + 32, 64, brow, kb);
      acc[j] = wmma_f16(af0, bf0, acc[j]);
      acc[j] = wmma_f16(af1, bf1, acc[j]);
    }
    __syncthreads();   // all reads of buf `cur` done before it is restaged
  }

#pragma unroll
  for (int j = 0; j < 2; ++j) {
    const int n = n0 + 32 * nw + 16 * j + l16;
    const float bv = bias[n];
#pragma unroll
    for (int vv = 0; vv < 8; ++vv) {
      const int m = m0 + 16 * mw + vv + 8 * lhalf;   // C layout: row = v + 8*(lane/16)
      const float val = (acc[j][vv] + bv) * outScale;
      if (mode == 2) {
        ((float*)outp)[(size_t)m * EMBED + n] = val;
      } else {
        const int bb = m >> 11, s = m & (SEQ - 1);
        const int hh = n >> 6,  d = n & (HSIZE - 1);
        const size_t o = (mode == 0)
            ? ((size_t)((bb * HEADS + hh) * SEQ + s)) * HSIZE + d
            : ((size_t)((bb * HEADS + hh) * HSIZE + d)) * SEQ + s;
        ((_Float16*)outp)[o] = (_Float16)val;
      }
    }
  }
}

// Fused flash attention. One wave per (b, h, 16-query tile); the 8 waves of a
// block share one (b,h), so K/V chunks are async-staged into LDS once per block,
// double-buffered so the DMA for chunk i+1 overlaps softmax+WMMA of chunk i.
// Qh,Kh: f16 [B,H,S,64]; Vt: f16 [B,H,64,S]; X out: f16 [B,S,EMBED]. Q pre-scaled.
__global__ __launch_bounds__(256)
void flash_attn(const _Float16* __restrict__ Qh, const _Float16* __restrict__ Kh,
                const _Float16* __restrict__ Vt, const int* __restrict__ mask,
                _Float16* __restrict__ X) {
  __shared__ _Float16 Ks[2][32 * 64];  // 32 keys x 64 d
  __shared__ _Float16 Vs[2][64 * 32];  // 64 d x 32 keys
  __shared__ _Float16 P[8][16 * 32];   // per-wave P staging (C-layout -> A-layout)

  const int tid   = threadIdx.x;
  const int wave  = tid >> 5;
  const int lane  = tid & 31;
  const int l16   = lane & 15;
  const int lhalf = lane >> 4;
  const int kb    = lhalf ? 8 : 0;

  const int g  = blockIdx.x * 8 + wave;   // global wave id, 3072 total
  const int bh = g >> 7;                  // same for all 8 waves of a block
  const int qt = g & 127;
  const int b  = bh / HEADS;
  const int h  = bh % HEADS;

  const _Float16* Qp = Qh + (size_t)bh * SEQ * HSIZE;
  const _Float16* Kp = Kh + (size_t)bh * SEQ * HSIZE;
  const _Float16* Vp = Vt + (size_t)bh * HSIZE * SEQ;
  const int* mp = mask + b * SEQ;

  // Staging assignment: K chunk = 32 rows x 8 segs, V chunk = 64 rows x 4 segs.
  const int krow = tid >> 3, kseg = (tid & 7) * 8;
  const int vrow = tid >> 2, vseg = (tid & 3) * 8;

  auto stage = [&](int kc, int buf) {
    async_b128(Kp + (kc + krow) * HSIZE + kseg, &Ks[buf][krow * 64 + kseg]);
    async_b128(Vp + vrow * SEQ + kc + vseg,     &Vs[buf][vrow * 32 + vseg]);
  };

  // Q fragments (row = qt*16 + l16, d in [0,32) and [32,64))
  const int qrow = qt * 16 + l16;
  const v16h qf0 = frag_ld(Qp,      HSIZE, qrow, kb);
  const v16h qf1 = frag_ld(Qp + 32, HSIZE, qrow, kb);

  float ml[8], ll[8];
  v8f oacc[4] = {{}, {}, {}, {}};
#pragma unroll
  for (int vv = 0; vv < 8; ++vv) { ml[vv] = -1e30f; ll[vv] = 0.f; }

  _Float16* Pl = P[wave];
  const int NCH = SEQ / 32;   // 64 chunks

  stage(0, 0);
#pragma unroll 1
  for (int it = 0; it < NCH; ++it) {
    const int cur = it & 1;
    const int kc  = it * 32;
    if (it + 1 < NCH) {
      stage((it + 1) * 32, cur ^ 1);   // next K/V chunk streams during compute
      wait_async_le<2>();
    } else {
      wait_async_le<0>();
    }
    __syncthreads();

    // ---- scores: 16 queries x 32 keys, f32 accum ----
    v8f s0 = {}, s1 = {};
    {
      v16h kf = frag_ld(Ks[cur],      64, l16, kb);        // keys 0..15, d 0..31
      s0 = wmma_f16(qf0, kf, s0);
      kf = frag_ld(Ks[cur] + 32, 64, l16, kb);             // d 32..63
      s0 = wmma_f16(qf1, kf, s0);
      kf = frag_ld(Ks[cur],      64, 16 + l16, kb);        // keys 16..31
      s1 = wmma_f16(qf0, kf, s1);
      kf = frag_ld(Ks[cur] + 32, 64, 16 + l16, kb);
      s1 = wmma_f16(qf1, kf, s1);
    }
    // ---- key mask (column = lane's n) ----
    const int mv0 = mp[kc + l16];
    const int mv1 = mp[kc + 16 + l16];
    if (mv0 == 0) {
#pragma unroll
      for (int vv = 0; vv < 8; ++vv) s0[vv] = -1e9f;
    }
    if (mv1 == 0) {
#pragma unroll
      for (int vv = 0; vv < 8; ++vv) s1[vv] = -1e9f;
    }
    // ---- online softmax: row stats via shfl within lane half ----
    float rm[8];
#pragma unroll
    for (int vv = 0; vv < 8; ++vv) rm[vv] = fmaxf(s0[vv], s1[vv]);
#pragma unroll
    for (int xm = 1; xm <= 8; xm <<= 1)
#pragma unroll
      for (int vv = 0; vv < 8; ++vv)
        rm[vv] = fmaxf(rm[vv], __shfl_xor(rm[vv], xm, 32));
    float corr[8], rs[8];
#pragma unroll
    for (int vv = 0; vv < 8; ++vv) {
      const float nm = fmaxf(ml[vv], rm[vv]);
      corr[vv] = __expf(ml[vv] - nm);
      ml[vv] = nm;
      const float p0 = __expf(s0[vv] - nm);
      const float p1 = __expf(s1[vv] - nm);
      s0[vv] = p0; s1[vv] = p1;
      rs[vv] = p0 + p1;
    }
#pragma unroll
    for (int xm = 1; xm <= 8; xm <<= 1)
#pragma unroll
      for (int vv = 0; vv < 8; ++vv)
        rs[vv] += __shfl_xor(rs[vv], xm, 32);
#pragma unroll
    for (int vv = 0; vv < 8; ++vv) ll[vv] = ll[vv] * corr[vv] + rs[vv];
#pragma unroll
    for (int t = 0; t < 4; ++t)
#pragma unroll
      for (int vv = 0; vv < 8; ++vv) oacc[t][vv] *= corr[vv];

    // ---- transpose P (C-layout) -> A-layout via per-wave LDS ----
#pragma unroll
    for (int vv = 0; vv < 8; ++vv) {
      const int r = vv + 8 * lhalf;
      Pl[r * 32 + l16]      = (_Float16)s0[vv];
      Pl[r * 32 + 16 + l16] = (_Float16)s1[vv];
    }
    asm volatile("s_wait_dscnt 0x0" ::: "memory");   // intra-wave LDS RAW (lane exchange)
    const v16h pf = frag_ld(Pl, 32, l16, kb);

    // ---- out += P(16x32) * V(32x64) ----
#pragma unroll
    for (int t = 0; t < 4; ++t) {
      const v16h vf = frag_ld(Vs[cur], 32, 16 * t + l16, kb);
      oacc[t] = wmma_f16(pf, vf, oacc[t]);
    }
    __syncthreads();   // all waves done with buf `cur` before it is restaged
  }

  // ---- epilogue: normalize, scatter f16 to X [B,S,EMBED] ----
  float inv[8];
#pragma unroll
  for (int vv = 0; vv < 8; ++vv) inv[vv] = 1.0f / ll[vv];
#pragma unroll
  for (int t = 0; t < 4; ++t) {
    const int e = h * HSIZE + 16 * t + l16;
#pragma unroll
    for (int vv = 0; vv < 8; ++vv) {
      const int s = qt * 16 + vv + 8 * lhalf;
      X[((size_t)(b * SEQ + s)) * EMBED + e] = (_Float16)(oacc[t][vv] * inv[vv]);
    }
  }
}

extern "C" void kernel_launch(void* const* d_in, const int* in_sizes, int n_in,
                              void* d_out, int out_size, void* d_ws, size_t ws_size,
                              hipStream_t stream) {
  const float* q    = (const float*)d_in[0];
  const float* k    = (const float*)d_in[1];
  const float* v    = (const float*)d_in[2];
  const int*   mask = (const int*)  d_in[3];
  const float* Wq   = (const float*)d_in[4];
  const float* bq   = (const float*)d_in[5];
  const float* Wk   = (const float*)d_in[6];
  const float* bk   = (const float*)d_in[7];
  const float* Wv   = (const float*)d_in[8];
  const float* bv   = (const float*)d_in[9];
  const float* Wo   = (const float*)d_in[10];
  const float* bo   = (const float*)d_in[11];

  const size_t actElems  = (size_t)MTOK * EMBED;     // 3,145,728
  const size_t wgtElems  = (size_t)EMBED * EMBED;    //   589,824
  const size_t headElems = (size_t)BATCH * HEADS * SEQ * HSIZE;  // 3,145,728

  _Float16* p = (_Float16*)d_ws;
  _Float16* q16  = p; p += actElems;
  _Float16* k16  = p; p += actElems;
  _Float16* v16  = p; p += actElems;
  _Float16* Wq16 = p; p += wgtElems;
  _Float16* Wk16 = p; p += wgtElems;
  _Float16* Wv16 = p; p += wgtElems;
  _Float16* Wo16 = p; p += wgtElems;
  _Float16* Qh   = p; p += headElems;
  _Float16* Kh   = p; p += headElems;
  _Float16* Vt   = p; p += headElems;
  _Float16* X16  = p; p += actElems;

  // f32 -> f16 precision downcast pass (read-once, HBM-bound).
  const int act4 = (int)(actElems / 4), wgt4 = (int)(wgtElems / 4);
  cvt_f16<<<(act4 + 255) / 256, 256, 0, stream>>>(q,  q16,  act4);
  cvt_f16<<<(act4 + 255) / 256, 256, 0, stream>>>(k,  k16,  act4);
  cvt_f16<<<(act4 + 255) / 256, 256, 0, stream>>>(v,  v16,  act4);
  cvt_f16<<<(wgt4 + 255) / 256, 256, 0, stream>>>(Wq, Wq16, wgt4);
  cvt_f16<<<(wgt4 + 255) / 256, 256, 0, stream>>>(Wk, Wk16, wgt4);
  cvt_f16<<<(wgt4 + 255) / 256, 256, 0, stream>>>(Wv, Wv16, wgt4);
  cvt_f16<<<(wgt4 + 255) / 256, 256, 0, stream>>>(Wo, Wo16, wgt4);

  const dim3 ggrid(EMBED / 64, MTOK / 64);   // (12, 64)
  const float qscale = 0.125f;               // 1/sqrt(HSIZE) folded into Q

  gemm_wmma<<<ggrid, 256, 0, stream>>>(q16, Wq16, bq, (void*)Qh, 0, qscale);
  gemm_wmma<<<ggrid, 256, 0, stream>>>(k16, Wk16, bk, (void*)Kh, 0, 1.0f);
  gemm_wmma<<<ggrid, 256, 0, stream>>>(v16, Wv16, bv, (void*)Vt, 1, 1.0f);

  const int totalWaves = BATCH * HEADS * (SEQ / 16);   // 3072
  flash_attn<<<totalWaves / 8, 256, 0, stream>>>(Qh, Kh, Vt, mask, X16);

  gemm_wmma<<<ggrid, 256, 0, stream>>>(X16, Wo16, bo, d_out, 2, 1.0f);
}
